// GAT2_57509612093519
// MI455X (gfx1250) — matrix operand order repaired and checked by reading, not compiled, not run on hardware
//
#include <hip/hip_runtime.h>
#include <hip/hip_bf16.h>

#define NNODES 50000
#define NEDGES 800000
#define EPLUS  (NEDGES + NNODES)

typedef __attribute__((ext_vector_type(16))) _Float16 v16h;
typedef __attribute__((ext_vector_type(8)))  float    v8f;

// ---- monotone float<->uint encoding for atomic segment-max ----
__device__ __forceinline__ unsigned fenc(float f) {
    unsigned u = __float_as_uint(f);
    return (u & 0x80000000u) ? ~u : (u | 0x80000000u);
}
__device__ __forceinline__ float fdec(unsigned u) {
    u = (u & 0x80000000u) ? (u & 0x7fffffffu) : ~u;
    return __uint_as_float(u);
}

// =====================================================================
// C[M x 64] = A[M x K] * W[K x 64] + bias   (f32 I/O)
// Split-f16 WMMA (Markidis): A = ah+al, B = bh+bl,
//   acc += ah*bh + al*bh + ah*bl   -> ~f32 precision, 3 WMMAs / k-step.
// 8 waves per block; each wave computes one 16x16 tile with K/32 steps.
// A fragment (16x32 f16, ISA 7.12.2): lane L holds row L&15;
//   elems 0..7 -> K = kb..kb+7, elems 8..15 -> K = kb+16..kb+23,
//   kb = (L&16)?8:0.
// B fragment (32x16 f16): lane L holds col N=L&15, elems 0..15 -> same K order as A.
// C (16x16 f32): vgpr r -> M = r + ((L&16)?8:0), N = L&15.
// =====================================================================
__global__ __launch_bounds__(256) void gemm_wmma_f16x3(
    const float* __restrict__ A, const float* __restrict__ W,
    const float* __restrict__ bias, float* __restrict__ C,
    int M, int K) {
    const int lane    = threadIdx.x & 31;
    const int wave    = threadIdx.x >> 5;
    const int rowtile = blockIdx.x * 2 + (wave >> 2);
    const int coltile = wave & 3;
    if (rowtile * 16 >= M) return;   // wave-uniform: EXEC stays all-1s

    v8f acc = {};
    const int m  = rowtile * 16 + (lane & 15);
    const int kb = (lane & 16) ? 8 : 0;
    const int ksteps = K >> 5;
    for (int kk = 0; kk < ksteps; ++kk) {
        const float* arow = A + (size_t)m * K + kk * 32;
        v16h ah, al, bh, bl;
#pragma unroll
        for (int e = 0; e < 8; ++e) {
            float v0 = arow[kb + e];
            float v1 = arow[kb + 16 + e];
            _Float16 h0 = (_Float16)v0;
            _Float16 h1 = (_Float16)v1;
            ah[e]     = h0;  al[e]     = (_Float16)(v0 - (float)h0);
            ah[e + 8] = h1;  al[e + 8] = (_Float16)(v1 - (float)h1);
        }
        const float* wrow = W + (size_t)(kk * 32) * 64 + coltile * 16 + (lane & 15);
#pragma unroll
        for (int e = 0; e < 16; ++e) {
            const int kr = (e < 8) ? (kb + e) : (kb + 8 + e);
            float v = wrow[(size_t)kr * 64];
            _Float16 hv = (_Float16)v;
            bh[e] = hv;
            bl[e] = (_Float16)(v - (float)hv);
        }
        acc = __builtin_amdgcn_wmma_f32_16x16x32_f16(
            false, ah, false, bh, (short)0, acc, false, false);
        acc = __builtin_amdgcn_wmma_f32_16x16x32_f16(
            false, al, false, bh, (short)0, acc, false, false);
        acc = __builtin_amdgcn_wmma_f32_16x16x32_f16(
            false, ah, false, bl, (short)0, acc, false, false);
    }
    const int   col = coltile * 16 + (lane & 15);
    const int   m0  = rowtile * 16 + ((lane & 16) ? 8 : 0);
    const float bv  = bias[col];
#pragma unroll
    for (int r = 0; r < 8; ++r)
        C[(size_t)(m0 + r) * 64 + col] = acc[r] + bv;
}

// ---------------------------------------------------------------------
// Pass 1: per (edge, head): logit = sum_d leakyrelu(xl[src]+xr[dst])*att
//         store logit; atomic segment-max into nmax[dst,h]
// ---------------------------------------------------------------------
template <int H, int D>
__global__ __launch_bounds__(256) void edge_logits(
    const int* __restrict__ ei, const float* __restrict__ xl,
    const float* __restrict__ xr, const float* __restrict__ att,
    float* __restrict__ elog, unsigned* __restrict__ nmax) {
    long long idx = (long long)blockIdx.x * blockDim.x + threadIdx.x;
    if (idx >= (long long)EPLUS * H) return;
    const int e = (int)(idx / H), h = (int)(idx % H);
    int src, dst;
    if (e < NEDGES) { src = ei[e]; dst = ei[NEDGES + e]; }
    else            { src = dst = e - NEDGES; }
    const float* pl = xl + (size_t)src * (H * D) + h * D;
    const float* pr = xr + (size_t)dst * (H * D) + h * D;
    const float* pa = att + h * D;
    float s = 0.f;
#pragma unroll
    for (int d = 0; d < D; ++d) {
        float mv = pl[d] + pr[d];
        mv = (mv > 0.f) ? mv : 0.2f * mv;
        s += mv * pa[d];
    }
    elog[(size_t)e * H + h] = s;
    atomicMax(&nmax[(size_t)dst * H + h], fenc(s));
}

// ---------------------------------------------------------------------
// Pass 2: ex = exp(logit - max[dst]); store; atomicAdd denom[dst,h]
// ---------------------------------------------------------------------
template <int H>
__global__ __launch_bounds__(256) void edge_exp(
    const int* __restrict__ ei, float* __restrict__ elog,
    const unsigned* __restrict__ nmax, float* __restrict__ denom) {
    long long idx = (long long)blockIdx.x * blockDim.x + threadIdx.x;
    if (idx >= (long long)EPLUS * H) return;
    const int e = (int)(idx / H), h = (int)(idx % H);
    int dst = (e < NEDGES) ? ei[NEDGES + e] : (e - NEDGES);
    const float mx = fdec(nmax[(size_t)dst * H + h]);
    const float ex = __expf(elog[(size_t)e * H + h] - mx);
    elog[(size_t)e * H + h] = ex;
    atomicAdd(&denom[(size_t)dst * H + h], ex);
}

// ---------------------------------------------------------------------
// Pass 3: out[dst, h*D+d] += xl[src, h*D+d] * (ex / denom[dst,h])
// one thread per (edge, channel)
// ---------------------------------------------------------------------
template <int H, int D>
__global__ __launch_bounds__(256) void edge_aggr(
    const int* __restrict__ ei, const float* __restrict__ xl,
    const float* __restrict__ elog, const float* __restrict__ denom,
    float* __restrict__ out) {
    long long idx = (long long)blockIdx.x * blockDim.x + threadIdx.x;
    if (idx >= (long long)EPLUS * (H * D)) return;
    const int e = (int)(idx / (H * D));
    const int c = (int)(idx % (H * D));
    const int h = c / D;
    int src, dst;
    if (e < NEDGES) { src = ei[e]; dst = ei[NEDGES + e]; }
    else            { src = dst = e - NEDGES; }
    const float alpha = elog[(size_t)e * H + h] / denom[(size_t)dst * H + h];
    atomicAdd(&out[(size_t)dst * (H * D) + c],
              xl[(size_t)src * (H * D) + c] * alpha);
}

// h[i] = elu(h[i] + bias[i%64])  (in place)
__global__ __launch_bounds__(256) void elu_bias64(
    float* __restrict__ h, const float* __restrict__ bias, long long total) {
    long long idx = (long long)blockIdx.x * blockDim.x + threadIdx.x;
    if (idx >= total) return;
    float v = h[idx] + bias[(int)(idx & 63)];
    h[idx] = (v > 0.f) ? v : (__expf(v) - 1.f);
}

// per row (wave32): v = elu(row + bias2); row = v - logsumexp(v)  (in place)
__global__ __launch_bounds__(256) void elu_logsoftmax64(
    float* __restrict__ out, const float* __restrict__ bias, int n) {
    const int wave = (int)((blockIdx.x * (long long)blockDim.x + threadIdx.x) >> 5);
    const int lane = threadIdx.x & 31;
    if (wave >= n) return;
    float* row = out + (size_t)wave * 64;
    float v0 = row[lane]      + bias[lane];
    float v1 = row[lane + 32] + bias[lane + 32];
    v0 = (v0 > 0.f) ? v0 : (__expf(v0) - 1.f);
    v1 = (v1 > 0.f) ? v1 : (__expf(v1) - 1.f);
    float m = fmaxf(v0, v1);
#pragma unroll
    for (int o = 16; o > 0; o >>= 1) m = fmaxf(m, __shfl_xor(m, o, 32));
    float s = __expf(v0 - m) + __expf(v1 - m);
#pragma unroll
    for (int o = 16; o > 0; o >>= 1) s += __shfl_xor(s, o, 32);
    const float lse = __logf(s) + m;
    row[lane]      = v0 - lse;
    row[lane + 32] = v1 - lse;
}

extern "C" void kernel_launch(void* const* d_in, const int* in_sizes, int n_in,
                              void* d_out, int out_size, void* d_ws, size_t ws_size,
                              hipStream_t stream) {
    const float* x     = (const float*)d_in[0];
    const int*   ei    = (const int*)  d_in[1];
    const float* W1l   = (const float*)d_in[2];
    const float* b1l   = (const float*)d_in[3];
    const float* W1r   = (const float*)d_in[4];
    const float* b1r   = (const float*)d_in[5];
    const float* att1  = (const float*)d_in[6];
    const float* bias1 = (const float*)d_in[7];
    const float* W2l   = (const float*)d_in[8];
    const float* b2l   = (const float*)d_in[9];
    const float* W2r   = (const float*)d_in[10];
    const float* b2r   = (const float*)d_in[11];
    const float* att2  = (const float*)d_in[12];
    const float* bias2 = (const float*)d_in[13];
    float* out = (float*)d_out;

    // workspace layout (bytes): xl | xr | h | elog | nmax | denom
    char* p = (char*)d_ws;
    const size_t nodeF = (size_t)NNODES * 64;
    float*    xl    = (float*)p;            p += nodeF * 4;
    float*    xr    = (float*)p;            p += nodeF * 4;
    float*    h     = (float*)p;            p += nodeF * 4;
    float*    elog  = (float*)p;            p += (size_t)EPLUS * 8 * 4;
    unsigned* nmax  = (unsigned*)p;         p += (size_t)NNODES * 8 * 4;
    float*    denom = (float*)p;            p += (size_t)NNODES * 8 * 4;

    const int TB = 256;
    const int rowtiles = (NNODES + 15) / 16;          // 3125
    const int gemmGrid = (rowtiles + 1) / 2;          // 2 rowtiles x 4 coltiles per block
    const long long eh1 = (long long)EPLUS * 8;
    const long long ec1 = (long long)EPLUS * 64;

    // ---------------- layer 1 ----------------
    gemm_wmma_f16x3<<<gemmGrid, TB, 0, stream>>>(x, W1l, b1l, xl, NNODES, 128);
    gemm_wmma_f16x3<<<gemmGrid, TB, 0, stream>>>(x, W1r, b1r, xr, NNODES, 128);

    hipMemsetAsync(nmax,  0, (size_t)NNODES * 8 * 4, stream);
    hipMemsetAsync(denom, 0, (size_t)NNODES * 8 * 4, stream);
    hipMemsetAsync(h,     0, nodeF * 4,              stream);

    edge_logits<8, 8><<<(int)((eh1 + TB - 1) / TB), TB, 0, stream>>>(ei, xl, xr, att1, elog, nmax);
    edge_exp<8>      <<<(int)((eh1 + TB - 1) / TB), TB, 0, stream>>>(ei, elog, nmax, denom);
    edge_aggr<8, 8>  <<<(int)((ec1 + TB - 1) / TB), TB, 0, stream>>>(ei, xl, elog, denom, h);
    elu_bias64       <<<(int)((nodeF + TB - 1) / TB), TB, 0, stream>>>(h, bias1, (long long)nodeF);

    // ---------------- layer 2 (H=1, D=64; aggregate straight into d_out) ---
    gemm_wmma_f16x3<<<gemmGrid, TB, 0, stream>>>(h, W2l, b2l, xl, NNODES, 64);
    gemm_wmma_f16x3<<<gemmGrid, TB, 0, stream>>>(h, W2r, b2r, xr, NNODES, 64);

    hipMemsetAsync(nmax,  0, (size_t)NNODES * 4, stream);
    hipMemsetAsync(denom, 0, (size_t)NNODES * 4, stream);
    hipMemsetAsync(out,   0, nodeF * 4,          stream);

    const long long eh2 = (long long)EPLUS;
    edge_logits<1, 64><<<(int)((eh2 + TB - 1) / TB), TB, 0, stream>>>(ei, xl, xr, att2, elog, nmax);
    edge_exp<1>       <<<(int)((eh2 + TB - 1) / TB), TB, 0, stream>>>(ei, elog, nmax, denom);
    edge_aggr<1, 64>  <<<(int)((ec1 + TB - 1) / TB), TB, 0, stream>>>(ei, xl, elog, denom, out);

    // elu + log_softmax in place on d_out: one wave32 per row
    elu_logsoftmax64<<<(NNODES * 32 + TB - 1) / TB, TB, 0, stream>>>(out, bias2, NNODES);
}
